// Codebook_12266426597621
// MI455X (gfx1250) — matrix operand order repaired and checked by reading, not compiled, not run on hardware
//
#include <hip/hip_runtime.h>

// ---------------------------------------------------------------------------
// Codebook nearest-neighbor search (vector quantization argmin) for gfx1250.
//   x: (8,4096,512) f32, codebook: (8192,512) f32 -> codes (8,4096)
// S = X @ C^T via v_wmma_f32_16x16x32_bf16 with hi/lo bf16 split (3 products,
// ~fp32 accuracy), fused argmin epilogue on 1/2||c||^2 - S.
// Codebook tiles double-buffered in LDS via GLOBAL_LOAD_ASYNC_TO_LDS_B128.
// Each buffer is a contiguous 64KB (hi|lo) so all DS offsets fit the 16-bit imm.
// ---------------------------------------------------------------------------

typedef __attribute__((ext_vector_type(16))) __bf16 v16bf;
typedef __attribute__((ext_vector_type(8)))  __bf16 v8bf;
typedef __attribute__((ext_vector_type(8)))  float  v8f;
typedef __attribute__((ext_vector_type(4)))  int    v4i;

constexpr int DDIM   = 512;      // feature dim (WMMA K total)
constexpr int NCODES = 8192;     // codebook entries
constexpr int MTOT   = 8 * 4096; // total points
constexpr int M_BLK  = 128;      // rows per workgroup (8 waves x 16)
constexpr int N_BLK  = 128;      // codes per tile
constexpr int KC     = 128;      // K chunk staged in LDS
constexpr int NCHUNK = DDIM / KC;                    // 4
constexpr int NSTEPS = (NCODES / N_BLK) * NCHUNK;    // 256 pipeline steps
constexpr int CHALF  = N_BLK * KC;                   // halves per (buf, hi|lo) plane

// pointer-to-addrspace types matching the async builtin's signature
typedef __attribute__((address_space(1))) v4i* gv4i_p;   // global int4*
typedef __attribute__((address_space(3))) v4i* lv4i_p;   // LDS int4*

// --- async 16B global -> LDS copy (per lane), tracked by ASYNCcnt ---
__device__ __forceinline__ void async_copy16(const __bf16* g, __bf16* l) {
#if __has_builtin(__builtin_amdgcn_global_load_async_to_lds_b128)
    __builtin_amdgcn_global_load_async_to_lds_b128(
        (gv4i_p)(uintptr_t)g,
        (lv4i_p)(uint32_t)(uintptr_t)l, 0, 0);
#else
    uint32_t lds = (uint32_t)(uintptr_t)l;
    asm volatile("global_load_async_to_lds_b128 %0, %1, off"
                 :: "v"(lds), "v"(g) : "memory");
#endif
}

template <int N> __device__ __forceinline__ void wait_asynccnt() {
#if __has_builtin(__builtin_amdgcn_s_wait_asynccnt)
    __builtin_amdgcn_s_wait_asynccnt(N);
#else
    asm volatile("s_wait_asynccnt %0" :: "i"(N) : "memory");
#endif
}

// --- split f32 -> bf16 hi + bf16 lo residual ---
__global__ __launch_bounds__(256) void split_bf16_kernel(
    const float* __restrict__ src, __bf16* __restrict__ hi,
    __bf16* __restrict__ lo, int n)
{
    int i = blockIdx.x * blockDim.x + threadIdx.x;
    if (i < n) {
        float f = src[i];
        __bf16 h = (__bf16)f;
        float r  = f - (float)h;
        hi[i] = h;
        lo[i] = (__bf16)r;
    }
}

// --- 0.5 * ||c_j||^2 in fp32 from the original codebook ---
__global__ __launch_bounds__(256) void cnorm_kernel(
    const float* __restrict__ cb, float* __restrict__ cn)
{
    int j = blockIdx.x * blockDim.x + threadIdx.x;
    if (j < NCODES) {
        const float* row = cb + (size_t)j * DDIM;
        float s = 0.0f;
        for (int k = 0; k < DDIM; ++k) s = fmaf(row[k], row[k], s);
        cn[j] = 0.5f * s;
    }
}

// --- main fused GEMM + argmin kernel ---
__global__ __launch_bounds__(256) void vq_argmin_kernel(
    const __bf16* __restrict__ xh, const __bf16* __restrict__ xl,
    const __bf16* __restrict__ ch, const __bf16* __restrict__ cl,
    const float*  __restrict__ cn, float* __restrict__ out)
{
    // double-buffered codebook chunk; each buffer contiguous: [hi 32KB | lo 32KB]
    __shared__ alignas(32) __bf16 sC[2][2 * CHALF];   // 128 KB total

    const int tid   = threadIdx.x;
    const int lane  = tid & 31;
    const int wave  = tid >> 5;
    const int l16   = lane & 15;
    const int hi16  = lane >> 4;            // 0 | 1
    const int khalfA = hi16 * 16;           // A frag: lanes>=16 hold K 16..31 run
    const int koffB  = hi16 * 16;           // B frag: lanes>=16 hold K 16..31 run

    const int mrow = blockIdx.x * M_BLK + wave * 16 + l16;   // A row for this lane
    const __bf16* __restrict__ xhrow = xh + (size_t)mrow * DDIM;
    const __bf16* __restrict__ xlrow = xl + (size_t)mrow * DDIM;

    // async-copy assignment: thread -> (code row, 64-half slice); 128B per plane
    const int ccode = tid >> 1;
    const int ck0   = (tid & 1) * (KC / 2);

    // issue async loads for pipeline step s into buffer (s & 1)
    auto issue_chunk = [&](int step) {
        const int nt2 = step >> 2;                     // code tile
        const int kc2 = (step & (NCHUNK - 1)) * KC;    // k chunk
        const int buf = step & 1;
        const __bf16* gh = ch + (size_t)(nt2 * N_BLK + ccode) * DDIM + kc2 + ck0;
        const __bf16* gl = cl + (size_t)(nt2 * N_BLK + ccode) * DDIM + kc2 + ck0;
        __bf16* dh = &sC[buf][ccode * KC + ck0];           // hi plane
        __bf16* dl = &sC[buf][CHALF + ccode * KC + ck0];   // lo plane
#pragma unroll
        for (int i = 0; i < (KC / 2) / 8; ++i) {       // 8 x 16B per plane
            async_copy16(gh + i * 8, dh + i * 8);
            async_copy16(gl + i * 8, dl + i * 8);
        }
    };

    float minval[8];
    int   minidx[8];
#pragma unroll
    for (int r = 0; r < 8; ++r) { minval[r] = 3.4e38f; minidx[r] = 0; }

    issue_chunk(0);   // prologue: chunk 0 in flight

    for (int nt = 0; nt < NCODES / N_BLK; ++nt) {
        const int cbase = nt * N_BLK;

        v8f acc[8];
#pragma unroll
        for (int s = 0; s < 8; ++s) acc[s] = 0;

        for (int ic = 0; ic < NCHUNK; ++ic) {
            const int step = nt * NCHUNK + ic;
            const int buf  = step & 1;
            const int kc   = ic * KC;

            wait_asynccnt<0>();   // this wave's loads for chunk `step` done
            __syncthreads();      // everyone's loads done, prev compute done
            if (step + 1 < NSTEPS) issue_chunk(step + 1);  // overlap next chunk

            // single base per chunk; hi at +0, lo at +CHALF (all imm-reachable)
            const __bf16* __restrict__ bbase = sC[buf];

#pragma unroll
            for (int w = 0; w < KC / 32; ++w) {
                const int kw = kc + w * 32;
                // A fragments (16x32 bf16): two 8-half runs per lane
                v8bf a0h = *(const v8bf*)(xhrow + kw + khalfA);
                v8bf a1h = *(const v8bf*)(xhrow + kw + 8 + khalfA);
                v8bf a0l = *(const v8bf*)(xlrow + kw + khalfA);
                v8bf a1l = *(const v8bf*)(xlrow + kw + 8 + khalfA);
                v16bf ah = __builtin_shufflevector(a0h, a1h, 0,1,2,3,4,5,6,7,8,9,10,11,12,13,14,15);
                v16bf al = __builtin_shufflevector(a0l, a1l, 0,1,2,3,4,5,6,7,8,9,10,11,12,13,14,15);
#pragma unroll
                for (int s = 0; s < 8; ++s) {
                    const int code = s * 16 + l16;  // B column for this lane
                    const v16bf bh = *(const v16bf*)(bbase + code * KC + w * 32 + koffB);
                    const v16bf bl = *(const v16bf*)(bbase + CHALF + code * KC + w * 32 + koffB);
                    // S += Ah*Bh + Ah*Bl + Al*Bh   (drop Al*Bl: ~2^-18 relative)
                    acc[s] = __builtin_amdgcn_wmma_f32_16x16x32_bf16(
                        false, ah, false, bh, (short)0, acc[s], false, false);
                    acc[s] = __builtin_amdgcn_wmma_f32_16x16x32_bf16(
                        false, ah, false, bl, (short)0, acc[s], false, false);
                    acc[s] = __builtin_amdgcn_wmma_f32_16x16x32_bf16(
                        false, al, false, bh, (short)0, acc[s], false, false);
                }
            }
        }

        // fused argmin epilogue: score = 0.5||c||^2 - x.c
#pragma unroll
        for (int s = 0; s < 8; ++s) {
            const int col = cbase + s * 16 + l16;
            const float c2 = cn[col];
#pragma unroll
            for (int r = 0; r < 8; ++r) {
                float score = c2 - acc[s][r];
                if (score < minval[r]) { minval[r] = score; minidx[r] = col; }
            }
        }
    }

    // cross-lane min+argmin over the 16 lanes of each half-wave
    // (C/D layout: VGPR r <-> row r + 8*(lane>=16), col = lane%16)
#pragma unroll
    for (int off = 1; off < 16; off <<= 1) {
#pragma unroll
        for (int r = 0; r < 8; ++r) {
            float ov = __shfl_xor(minval[r], off, 32);
            int   oi = __shfl_xor(minidx[r], off, 32);
            if (ov < minval[r] || (ov == minval[r] && oi < minidx[r])) {
                minval[r] = ov; minidx[r] = oi;
            }
        }
    }
    if (l16 == 0) {
        const int rowbase = blockIdx.x * M_BLK + wave * 16 + hi16 * 8;
#pragma unroll
        for (int r = 0; r < 8; ++r) out[rowbase + r] = (float)minidx[r];
    }
}

extern "C" void kernel_launch(void* const* d_in, const int* in_sizes, int n_in,
                              void* d_out, int out_size, void* d_ws, size_t ws_size,
                              hipStream_t stream) {
    const float* x  = (const float*)d_in[0];   // (8,4096,512) f32
    const float* cb = (const float*)d_in[1];   // (8192,512) f32
    float* out = (float*)d_out;                // (8,4096) codes

    const size_t XN = (size_t)MTOT * DDIM;     // 16.7M
    const size_t CN = (size_t)NCODES * DDIM;   // 4.2M

    char* ws = (char*)d_ws;
    __bf16* xh = (__bf16*)ws;                                 // 32 MB
    __bf16* xl = xh + XN;                                     // 32 MB
    __bf16* ch = (__bf16*)(ws + 2 * XN * sizeof(__bf16));     // 8 MB
    __bf16* cl = ch + CN;                                     // 8 MB
    float*  cn = (float*)(ws + 2 * XN * sizeof(__bf16) + 2 * CN * sizeof(__bf16));

    split_bf16_kernel<<<(int)((XN + 255) / 256), 256, 0, stream>>>(x,  xh, xl, (int)XN);
    split_bf16_kernel<<<(int)((CN + 255) / 256), 256, 0, stream>>>(cb, ch, cl, (int)CN);
    cnorm_kernel<<<(NCODES + 255) / 256, 256, 0, stream>>>(cb, cn);

    vq_argmin_kernel<<<MTOT / M_BLK, 256, 0, stream>>>(xh, xl, ch, cl, cn, out);
}
